// MultiHeadAttentionBlockRepMode_28784870818214
// MI455X (gfx1250) — matrix operand order repaired and checked by reading, not compiled, not run on hardware
//
#include <hip/hip_runtime.h>
#include <hip/hip_bf16.h>

#define D_MODEL 1024
#define N_HEADS 16
#define D_K     64
#define LL      1024
#define LP      (LL + 4)        // padded rows per batch (2 zero rows each side)
#define BBATCH  4
#define M_TOTAL (BBATCH * LL)   // 4096

typedef __attribute__((ext_vector_type(16))) _Float16 v16h;
typedef __attribute__((ext_vector_type(8)))  _Float16 v8h;
typedef __attribute__((ext_vector_type(8)))  float    v8f;

#define WMMA_F16(A, B, C) \
  __builtin_amdgcn_wmma_f32_16x16x32_f16(false, (A), false, (B), (short)0, (C), false, false)

__device__ inline v8f zero8f() {
  v8f z;
#pragma unroll
  for (int i = 0; i < 8; ++i) z[i] = 0.0f;
  return z;
}

// A-matrix 16x32 f16 fragment (ISA 7.12.2): lane holds row M=lane&15.
// v16h elements 0..7  = K in [k0+8*half,    k0+8*half+8)
// v16h elements 8..15 = K in [k0+16+8*half, k0+16+8*half+8)
__device__ inline v16h load_a16(const _Float16* __restrict__ row, int k0, int half) {
  union { v16h v; v8h h2[2]; } u;
  u.h2[0] = *(const v8h*)(row + k0 + 8 * half);
  u.h2[1] = *(const v8h*)(row + k0 + 16 + 8 * half);
  return u.v;
}

// B-matrix 32x16 f16 fragment: lane holds col N=lane&15; elements 0..7 K=8*half..+8,
// elements 8..15 K=16+8*half..+8. `row` points at the K-contiguous column of B.
__device__ inline v16h load_b16(const _Float16* __restrict__ row, int k0, int half) {
  union { v16h v; v8h h2[2]; } u;
  u.h2[0] = *(const v8h*)(row + k0 + 8 * half);
  u.h2[1] = *(const v8h*)(row + k0 + 16 + 8 * half);
  return u.v;
}

// ---------------------------------------------------------------- gate softmax
__global__ void gate_kernel(const float* __restrict__ gw, const float* __restrict__ gb,
                            float* __restrict__ G) {
  const int i = blockIdx.x;       // projection 0..3
  const int o = threadIdx.x;      // channel 0..1023
  float e[5];
  float mx = -1e30f;
#pragma unroll
  for (int E = 0; E < 5; ++E) {
    e[E] = gw[i * 5120 + E * 1024 + o] + gb[i * 5120 + E * 1024 + o];
    mx = fmaxf(mx, e[E]);
  }
  float s = 0.f;
#pragma unroll
  for (int E = 0; E < 5; ++E) { e[E] = __expf(e[E] - mx); s += e[E]; }
  const float inv = 1.f / s;
#pragma unroll
  for (int E = 0; E < 5; ++E) G[((i << 10) + o) * 5 + E] = e[E] * inv;
}

// ------------------------------------------------- fold 5 experts -> 5-tap f16
// Wc layout: [proj][tap][o][c] f16  (o-major rows => B fragments contiguous)
__global__ void combine_kernel(const float* __restrict__ c5, const float* __restrict__ c3,
                               const float* __restrict__ c1, const float* __restrict__ a3,
                               const float* __restrict__ a5, const float* __restrict__ G,
                               _Float16* __restrict__ Wc) {
  const int idx = blockIdx.x * 256 + threadIdx.x;   // 4M threads
  const int i = idx >> 20;
  const int rem = idx & 0xFFFFF;
  const int o = rem >> 10;
  const int c = rem & 1023;
  const size_t oc = ((size_t)i << 20) + ((size_t)o << 10) + (size_t)c;

  const float* g = G + (((size_t)i << 10) + o) * 5;
  const float g0 = g[0], g1 = g[1], g2 = g[2], g3 = g[3], g4 = g[4];

  const float a3c = a3[oc] * (g3 * (1.0f / 3.0f));
  const float a5c = a5[oc] * (g4 * 0.2f);
  const float k1c = c1[oc] * g2;

  float w[5];
  w[0] = g0 * c5[oc * 5 + 0]                       + a5c;
  w[1] = g0 * c5[oc * 5 + 1] + g1 * c3[oc * 3 + 0] + a3c + a5c;
  w[2] = g0 * c5[oc * 5 + 2] + g1 * c3[oc * 3 + 1] + k1c + a3c + a5c;
  w[3] = g0 * c5[oc * 5 + 3] + g1 * c3[oc * 3 + 2] + a3c + a5c;
  w[4] = g0 * c5[oc * 5 + 4]                       + a5c;

#pragma unroll
  for (int t = 0; t < 5; ++t)
    Wc[(((size_t)(i * 5 + t)) << 20) + ((size_t)o << 10) + c] = (_Float16)w[t];
}

// -------------------------------- f32 -> f16 with 2-row zero padding per batch
// dst layout [b][1028][1024]; rows 0,1 and 1026,1027 of each batch are zero.
__global__ void f32_to_f16_pad_kernel(const float* __restrict__ src,
                                      _Float16* __restrict__ dst) {
  const int idx = blockIdx.x * 256 + threadIdx.x;   // 1028*1024 per batch (exact)
  const int b = blockIdx.y;
  const int l = idx >> 10;    // 0..1027
  const int c = idx & 1023;
  float v = 0.f;
  if (l >= 2 && l < LL + 2) v = src[((size_t)(b * LL + l - 2) << 10) + c];
  dst[((size_t)(b * LP + l) << 10) + c] = (_Float16)v;
}

// zero the 4 pad rows per batch of a padded f16 buffer (for Ctx)
__global__ void zero_pad_kernel(_Float16* __restrict__ Xp) {
  const int idx = blockIdx.x * 256 + threadIdx.x;   // 4*4*1024 = 16384
  const int b = idx >> 12;
  const int r = (idx >> 10) & 3;
  const int c = idx & 1023;
  const int row = b * LP + (r < 2 ? r : LL + r);    // 0,1,1026,1027
  Xp[((size_t)row << 10) + c] = (_Float16)0;
}

// ------------------------------------------------------- conv-as-GEMM (WMMA)
// y[m,o] = sum_t sum_c Xp[b, l+t, c] * W[t][o][c]  (padded => branchless)
// One wave computes a 32(M) x 64(N) tile: 8 WMMA per 32-K chunk.
// mode 0: relu, f16 natural [m][o]   (Q,K)
// mode 1: relu, f16 per-head transposed Vt[(b*1024+o)][l]  (V)
// mode 2: no relu, fp32 natural -> d_out
__global__ __launch_bounds__(32)
void conv_gemm_kernel(const _Float16* __restrict__ Xp, const _Float16* __restrict__ W5,
                      _Float16* __restrict__ outh, float* __restrict__ outf, int mode) {
  const int lane = threadIdx.x;
  const int n = lane & 15, half = lane >> 4;
  const int m0 = blockIdx.x * 32;
  const int n0 = blockIdx.y * 64;

  v8f acc[2][4];
#pragma unroll
  for (int mt = 0; mt < 2; ++mt)
#pragma unroll
    for (int nt = 0; nt < 4; ++nt) acc[mt][nt] = zero8f();

  // lane's A rows in padded space (before tap shift): b*1028 + l
  int abase[2];
#pragma unroll
  for (int mt = 0; mt < 2; ++mt) {
    const int row = m0 + mt * 16 + n;
    abase[mt] = (row >> 10) * LP + (row & 1023);
  }

#pragma unroll
  for (int t = 0; t < 5; ++t) {
    const _Float16* Wt = W5 + ((size_t)t << 20);
    const _Float16* arow0 = Xp + ((size_t)(abase[0] + t) << 10);
    const _Float16* arow1 = Xp + ((size_t)(abase[1] + t) << 10);
    const _Float16* brow[4];
#pragma unroll
    for (int nt = 0; nt < 4; ++nt)
      brow[nt] = Wt + ((size_t)(n0 + nt * 16 + n) << 10);

#pragma unroll 2
    for (int kc = 0; kc < D_MODEL; kc += 32) {
      // branchless in-bounds prefetch of the weight stream (global_prefetch_b8)
      __builtin_prefetch(brow[0] + ((kc + 128) & 1023), 0, 1);
      const v16h a0 = load_a16(arow0, kc, half);
      const v16h a1 = load_a16(arow1, kc, half);
#pragma unroll
      for (int nt = 0; nt < 4; ++nt) {
        const v16h bf = load_b16(brow[nt], kc, half);
        acc[0][nt] = WMMA_F16(a0, bf, acc[0][nt]);
        acc[1][nt] = WMMA_F16(a1, bf, acc[1][nt]);
      }
    }
  }

  // epilogue: C/D layout -> lane holds col n, rows v + 8*half of each 16x16 tile
#pragma unroll
  for (int mt = 0; mt < 2; ++mt) {
#pragma unroll
    for (int v = 0; v < 8; ++v) {
      const int row = m0 + mt * 16 + v + 8 * half;   // unpadded output row
#pragma unroll
      for (int nt = 0; nt < 4; ++nt) {
        const int col = n0 + nt * 16 + n;
        float val = acc[mt][nt][v];
        if (mode == 2) {
          outf[((size_t)row << 10) + col] = val;
        } else {
          val = fmaxf(val, 0.0f);
          if (mode == 0) {
            outh[((size_t)row << 10) + col] = (_Float16)val;
          } else {  // Vt[(b*1024 + col)][l] ; (b*16+h)*64+d == b*1024+col
            const int b = row >> 10, l = row & 1023;
            outh[(((size_t)(b * 1024 + col)) << 10) + l] = (_Float16)val;
          }
        }
      }
    }
  }
}

// ----------------------------------------------------- flash attention (WMMA)
// One wave per (b,h, 16-row q-tile). S = Q K^T / 8 masked, online softmax,
// O += P V with P bounced through LDS (C/D layout -> A layout re-shape).
__global__ __launch_bounds__(32)
void attn_kernel(const _Float16* __restrict__ Qf, const _Float16* __restrict__ Kf,
                 const _Float16* __restrict__ Vt, const int* __restrict__ mask,
                 _Float16* __restrict__ ctx /* padded layout */) {
  __shared__ _Float16 P[16 * 32];

  const int lane = threadIdx.x;
  const int n = lane & 15, half = lane >> 4;
  const int bh = blockIdx.x, b = bh >> 4, h = bh & 15;
  const int q0 = blockIdx.y << 4;

  const _Float16* qrow = Qf + ((size_t)(b * LL + q0 + n) << 10) + h * D_K;
  const v16h aQ0 = load_a16(qrow, 0, half);
  const v16h aQ1 = load_a16(qrow, 32, half);

  const float NEGMIN = -3.4028235e38f;  // finfo(f32).min
  float rm[8], rs[8];
  v8f O[4];
#pragma unroll
  for (int v = 0; v < 8; ++v) { rm[v] = NEGMIN; rs[v] = 0.f; }
#pragma unroll
  for (int j = 0; j < 4; ++j) O[j] = zero8f();

  for (int key0 = 0; key0 < LL; key0 += 32) {
    // ---- S tile 16x32 via 4 WMMAs (K-dim = dk = 64 in two chunks)
    v8f s0 = zero8f(), s1 = zero8f();
    {
      const _Float16* kr0 = Kf + ((size_t)(b * LL + key0 + n) << 10) + h * D_K;
      s0 = WMMA_F16(aQ0, load_b16(kr0, 0, half), s0);
      s0 = WMMA_F16(aQ1, load_b16(kr0, 32, half), s0);
      const _Float16* kr1 = Kf + ((size_t)(b * LL + key0 + 16 + n) << 10) + h * D_K;
      s1 = WMMA_F16(aQ0, load_b16(kr1, 0, half), s1);
      s1 = WMMA_F16(aQ1, load_b16(kr1, 32, half), s1);
    }
    // ---- scale + mask + chunk row-max (rows live in 16-lane groups)
    float mx[8];
#pragma unroll
    for (int v = 0; v < 8; ++v) {
      const int row = q0 + v + 8 * half;
      const int* mp = mask + ((size_t)(b * LL + row) << 10) + key0 + n;
      float e0 = s0[v] * 0.125f; if (mp[0] == 0) e0 = NEGMIN;
      float e1 = s1[v] * 0.125f; if (mp[16] == 0) e1 = NEGMIN;
      s0[v] = e0; s1[v] = e1;
      float m = fmaxf(e0, e1);
      m = fmaxf(m, __shfl_xor(m, 1));
      m = fmaxf(m, __shfl_xor(m, 2));
      m = fmaxf(m, __shfl_xor(m, 4));
      m = fmaxf(m, __shfl_xor(m, 8));
      mx[v] = m;
    }
    // ---- online softmax update + stage P in LDS (row-major 16x32 f16)
#pragma unroll
    for (int v = 0; v < 8; ++v) {
      const float nm = fmaxf(rm[v], mx[v]);
      const float al = __expf(rm[v] - nm);
      const float p0 = __expf(s0[v] - nm);
      const float p1 = __expf(s1[v] - nm);
      rm[v] = nm;
      float ls = p0 + p1;
      ls += __shfl_xor(ls, 1);
      ls += __shfl_xor(ls, 2);
      ls += __shfl_xor(ls, 4);
      ls += __shfl_xor(ls, 8);
      rs[v] = rs[v] * al + ls;
      O[0][v] *= al; O[1][v] *= al; O[2][v] *= al; O[3][v] *= al;
      const int prow = v + 8 * half;
      P[prow * 32 + n]      = (_Float16)p0;
      P[prow * 32 + 16 + n] = (_Float16)p1;
    }
    __syncthreads();
    // ---- re-shape P (C/D layout in LDS) into an A fragment
    v16h aP;
    {
      union { v16h v; v8h h2[2]; } u;
      const _Float16* pr = &P[(lane & 15) * 32];
      u.h2[0] = *(const v8h*)(pr + 8 * half);
      u.h2[1] = *(const v8h*)(pr + 16 + 8 * half);
      aP = u.v;
    }
    // ---- O(16x64) += P(16x32) x V(32x64) : 4 WMMAs, V from transposed layout
#pragma unroll
    for (int nt2 = 0; nt2 < 4; ++nt2) {
      const _Float16* vrow = Vt + ((size_t)(bh * D_K + nt2 * 16 + n) << 10) + key0;
      O[nt2] = WMMA_F16(aP, load_b16(vrow, 0, half), O[nt2]);
    }
    __syncthreads();
  }

  // ---- normalize and store context into PADDED [b][l+2][h*64+d] f16
#pragma unroll
  for (int v = 0; v < 8; ++v) {
    const float inv = 1.f / rs[v];
    const int l = q0 + v + 8 * half;
    _Float16* crow = ctx + ((size_t)(b * LP + 2 + l) << 10) + h * D_K + n;
    crow[0]  = (_Float16)(O[0][v] * inv);
    crow[16] = (_Float16)(O[1][v] * inv);
    crow[32] = (_Float16)(O[2][v] * inv);
    crow[48] = (_Float16)(O[3][v] * inv);
  }
}

// -----------------------------------------------------------------------------
extern "C" void kernel_launch(void* const* d_in, const int* in_sizes, int n_in,
                              void* d_out, int out_size, void* d_ws, size_t ws_size,
                              hipStream_t stream) {
  (void)in_sizes; (void)n_in; (void)out_size; (void)ws_size;

  const float* q   = (const float*)d_in[0];
  const float* k   = (const float*)d_in[1];
  const float* v   = (const float*)d_in[2];
  const float* gw  = (const float*)d_in[3];
  const float* gb  = (const float*)d_in[4];
  const float* c5  = (const float*)d_in[5];
  const float* c3  = (const float*)d_in[6];
  const float* c1  = (const float*)d_in[7];
  const float* a3  = (const float*)d_in[8];
  const float* a5  = (const float*)d_in[9];
  const int*   msk = (const int*)d_in[10];
  float* out = (float*)d_out;

  char* ws = (char*)d_ws;
  const size_t ACT  = (size_t)M_TOTAL * D_MODEL;            // 4096*1024
  const size_t ACTP = (size_t)BBATCH * LP * D_MODEL;        // 4*1028*1024 (padded)
  float*    G   = (float*)ws;                               // 80KB
  _Float16* Wc  = (_Float16*)(ws + (128 << 10));            // 40MB
  _Float16* Xq  = (_Float16*)(ws + (128 << 10) + (size_t)40 * (1 << 20));
  _Float16* Xk  = Xq + ACTP;
  _Float16* Xv  = Xk + ACTP;
  _Float16* Ctx = Xv + ACTP;                                // padded
  _Float16* Qf  = Ctx + ACTP;
  _Float16* Kf  = Qf + ACT;
  _Float16* Vt  = Kf + ACT;                                 // [b][h][dk][L]
  // total ~ 0.125 + 40 + 4*8.03 + 3*8 = ~96.3 MB

  gate_kernel<<<4, 1024, 0, stream>>>(gw, gb, G);
  combine_kernel<<<(4 * 1024 * 1024) / 256, 256, 0, stream>>>(c5, c3, c1, a3, a5, G, Wc);

  dim3 pgrid((LP * D_MODEL) / 256, BBATCH);                 // 4112 x 4, exact
  f32_to_f16_pad_kernel<<<pgrid, 256, 0, stream>>>(q, Xq);
  f32_to_f16_pad_kernel<<<pgrid, 256, 0, stream>>>(k, Xk);
  f32_to_f16_pad_kernel<<<pgrid, 256, 0, stream>>>(v, Xv);
  zero_pad_kernel<<<(BBATCH * 4 * D_MODEL) / 256, 256, 0, stream>>>(Ctx);

  const size_t PW = (size_t)5 << 20;   // elements per projection's folded weights
  dim3 cgrid(M_TOTAL / 32, D_MODEL / 64);
  conv_gemm_kernel<<<cgrid, 32, 0, stream>>>(Xq, Wc + 0 * PW, Qf, nullptr, 0);
  conv_gemm_kernel<<<cgrid, 32, 0, stream>>>(Xk, Wc + 1 * PW, Kf, nullptr, 0);
  conv_gemm_kernel<<<cgrid, 32, 0, stream>>>(Xv, Wc + 2 * PW, Vt, nullptr, 1);

  attn_kernel<<<dim3(BBATCH * N_HEADS, LL / 16), 32, 0, stream>>>(Qf, Kf, Vt, msk, Ctx);

  conv_gemm_kernel<<<cgrid, 32, 0, stream>>>(Ctx, Wc + 3 * PW, nullptr, out, 2);
}
